// LlamaLMHead_44796508897406
// MI455X (gfx1250) — matrix operand order repaired and checked by reading, not compiled, not run on hardware
//
#include <hip/hip_runtime.h>
#include <hip/hip_bf16.h>
#include <math.h>

#define B_ 64
#define H_ 4096
#define V_ 128000
#define RMS_EPS 1e-5f
#define TOPK 8

typedef __attribute__((ext_vector_type(16))) __bf16 v16bf;
typedef __attribute__((ext_vector_type(2)))  __bf16 v2bf;
typedef __attribute__((ext_vector_type(2)))  float  v2f;
typedef __attribute__((ext_vector_type(8)))  float  v8f;
typedef int v4i_vs __attribute__((vector_size(4 * sizeof(int))));

#define AS_GLOBAL __attribute__((address_space(1)))
#define AS_LDS    __attribute__((address_space(3)))

union Frag {
    v16bf    v;
    unsigned u[8];
    uint4    q[2];
};

// ---------- bf16 helpers ----------
// two fp32 -> packed bf16 pair (RNE). Prefer the dedicated builtin; otherwise
// a vector fptrunc, which selects v_cvt_pk_bf16_f32 when the target has it.
__device__ __forceinline__ unsigned cvt_pk_bf16(float a, float b) {
#if defined(__has_builtin) && __has_builtin(__builtin_amdgcn_cvt_pk_bf16_f32)
    v2bf p = __builtin_amdgcn_cvt_pk_bf16_f32(a, b);
    return __builtin_bit_cast(unsigned, p);
#else
    v2f t;
    t.x = a;
    t.y = b;
    v2bf p = __builtin_convertvector(t, v2bf);
    return __builtin_bit_cast(unsigned, p);
#endif
}
__device__ __forceinline__ float lo_f(unsigned pk) {   // first element as fp32
    return __builtin_bit_cast(float, pk << 16);
}
__device__ __forceinline__ float hi_f(unsigned pk) {   // second element as fp32
    return __builtin_bit_cast(float, pk & 0xffff0000u);
}
// (hi_pair, lo_pair) split of two fp32 values
__device__ __forceinline__ void split_pair(float a, float b, unsigned& hp, unsigned& lp) {
    hp = cvt_pk_bf16(a, b);
    lp = cvt_pk_bf16(a - lo_f(hp), b - hi_f(hp));
}

// ---------- async global->LDS copy (16B per lane), ASYNCcnt-tracked ----------
#if defined(__has_builtin) && __has_builtin(__builtin_amdgcn_global_load_async_to_lds_b128)
#define HAVE_ASYNC_LDS 1
__device__ __forceinline__ void async_copy16(const void* g, void* l) {
    __builtin_amdgcn_global_load_async_to_lds_b128(
        (AS_GLOBAL v4i_vs*)(uintptr_t)g,
        (AS_LDS v4i_vs*)(unsigned)(uintptr_t)l, 0, 0);
}
__device__ __forceinline__ void async_wait0() {
#if __has_builtin(__builtin_amdgcn_s_wait_asynccnt)
    __builtin_amdgcn_s_wait_asynccnt(0);
#else
    asm volatile("s_wait_asynccnt 0x0" ::: "memory");
#endif
}
#else
#define HAVE_ASYNC_LDS 0
__device__ __forceinline__ void async_wait0() {}
#endif

// stage one 256-wide K chunk of packed activations into an LDS buffer
__device__ __forceinline__ void stage_chunk(uint4* dst_hi, uint4* dst_lo,
                                            const uint4* gh, const uint4* gl,
                                            int kc, int tid) {
    for (int i = tid; i < 2048; i += 256) {
        int row  = i >> 5;
        int c4   = i & 31;
        int gidx = row * 512 + (kc >> 3) + c4;
#if HAVE_ASYNC_LDS
        async_copy16(&gh[gidx], &dst_hi[i]);
        async_copy16(&gl[gidx], &dst_lo[i]);
#else
        dst_hi[i] = gh[gidx];
        dst_lo[i] = gl[gidx];
#endif
    }
}

// ---------- sortable (value, index) key: max value, ties -> lowest index ----------
__device__ __forceinline__ unsigned long long argmax_key(float f, int idx) {
    unsigned u = __builtin_bit_cast(unsigned, f);
    unsigned k = (u & 0x80000000u) ? ~u : (u | 0x80000000u);
    return ((unsigned long long)k << 32) | (unsigned)(~idx);
}
__device__ __forceinline__ float key_val(unsigned long long k) {
    unsigned ku = (unsigned)(k >> 32);
    unsigned s  = (ku & 0x80000000u) ? (ku & 0x7fffffffu) : ~ku;
    return __builtin_bit_cast(float, s);
}
__device__ __forceinline__ int key_idx(unsigned long long k) {
    return (int)(~(unsigned)(k & 0xffffffffu));
}

// =====================================================================
// Kernel 1: RMSNorm + fp32 -> (bf16_hi, bf16_lo) split, packed pairs.
// a_hi/a_lo: [64][2048] uints (2 bf16 per uint, K-major).
// =====================================================================
__global__ __launch_bounds__(256) void rmsnorm_split_kernel(
    const float* __restrict__ x, const float* __restrict__ w,
    unsigned* __restrict__ a_hi, unsigned* __restrict__ a_lo) {
    int b = blockIdx.x, t = threadIdx.x;
    const float4* xr = (const float4*)(x + (size_t)b * H_);
    const float4* wr = (const float4*)w;

    float4 xv[4], wv[4];
    float ss = 0.f;
#pragma unroll
    for (int i = 0; i < 4; ++i) {
        int i4 = t + i * 256;
        xv[i] = xr[i4];
        wv[i] = wr[i4];
        ss += xv[i].x * xv[i].x + xv[i].y * xv[i].y +
              xv[i].z * xv[i].z + xv[i].w * xv[i].w;
    }
    __shared__ float red[256];
    red[t] = ss;
    __syncthreads();
    for (int off = 128; off > 0; off >>= 1) {
        if (t < off) red[t] += red[t + off];
        __syncthreads();
    }
    float scale = rsqrtf(red[0] * (1.0f / H_) + RMS_EPS);

#pragma unroll
    for (int i = 0; i < 4; ++i) {
        int i4 = t + i * 256;
        float h0 = xv[i].x * scale * wv[i].x;
        float h1 = xv[i].y * scale * wv[i].y;
        float h2 = xv[i].z * scale * wv[i].z;
        float h3 = xv[i].w * scale * wv[i].w;
        unsigned hp0, lp0, hp1, lp1;
        split_pair(h0, h1, hp0, lp0);
        split_pair(h2, h3, hp1, lp1);
        size_t base = (size_t)b * (H_ / 2) + (size_t)i4 * 2;
        a_hi[base + 0] = hp0;
        a_hi[base + 1] = hp1;
        a_lo[base + 0] = lp0;
        a_lo[base + 1] = lp1;
    }
}

// =====================================================================
// Kernel 2: logits[64][128000] = hnorm x W^T via bf16 hi/lo split WMMA.
// Grid: V/128 workgroups x 256 threads (8 waves).
// Wave w computes the full M=64 x N=16 strip at columns n0 + 16*w.
// Activations double-buffered in LDS (2 x 64KB) via async-to-LDS copies:
//   issue(next) -> compute(cur) -> s_wait_asynccnt 0 -> barrier -> swap.
// =====================================================================
__global__ __launch_bounds__(256) void lmhead_gemm_wmma(
    const unsigned* __restrict__ a_hi, const unsigned* __restrict__ a_lo,
    const float* __restrict__ w, float* __restrict__ logits) {
    __shared__ uint4 sh_hi[2][2048];   // 64 rows x 128 uints (packed bf16 pairs)
    __shared__ uint4 sh_lo[2][2048];

    const int tid   = threadIdx.x;
    const int wave  = tid >> 5;
    const int lane  = tid & 31;
    const int n0    = blockIdx.x * 128;
    const int ncol  = n0 + wave * 16 + (lane & 15);
    const int khalf = lane >> 4;                   // which 16-wide K half this lane owns

    const float* wrow = w + (size_t)ncol * H_ + (size_t)khalf * 16;

    v8f acc[4];
#pragma unroll
    for (int mt = 0; mt < 4; ++mt)
#pragma unroll
        for (int r = 0; r < 8; ++r) acc[mt][r] = 0.0f;

    const uint4* gh = (const uint4*)a_hi;   // row stride = 512 uint4
    const uint4* gl = (const uint4*)a_lo;

    // prologue: fill buffer 0
    stage_chunk(sh_hi[0], sh_lo[0], gh, gl, 0, tid);
    async_wait0();
    __syncthreads();

    int buf = 0;
    for (int kc = 0; kc < H_; kc += 256) {
        if (kc + 256 < H_) {
            stage_chunk(sh_hi[buf ^ 1], sh_lo[buf ^ 1], gh, gl, kc + 256, tid);
            __builtin_prefetch(wrow + kc + 256, 0, 0);   // next weight chunk
        }
        const uint4* cur_hi = sh_hi[buf];
        const uint4* cur_lo = sh_lo[buf];

#pragma unroll
        for (int k2 = 0; k2 < 256; k2 += 32) {
            // ---- B fragment: lane reads 16 contiguous fp32 of its W row ----
            const float4* wp = (const float4*)(wrow + kc + k2);
            float4 f0 = wp[0], f1 = wp[1], f2 = wp[2], f3 = wp[3];
            float f[16] = { f0.x, f0.y, f0.z, f0.w, f1.x, f1.y, f1.z, f1.w,
                            f2.x, f2.y, f2.z, f2.w, f3.x, f3.y, f3.z, f3.w };
            Frag bhi, blo;
#pragma unroll
            for (int j = 0; j < 8; ++j)
                split_pair(f[2 * j], f[2 * j + 1], bhi.u[j], blo.u[j]);

            // ---- A fragments from LDS + 3 WMMAs per M tile (hi/lo split) ----
#pragma unroll
            for (int mt = 0; mt < 4; ++mt) {
                int m     = mt * 16 + (lane & 15);
                int base4 = m * 32 + (k2 >> 3) + 2 * khalf;
                Frag ahi, alo;
                ahi.q[0] = cur_hi[base4];
                ahi.q[1] = cur_hi[base4 + 1];
                alo.q[0] = cur_lo[base4];
                alo.q[1] = cur_lo[base4 + 1];
                acc[mt] = __builtin_amdgcn_wmma_f32_16x16x32_bf16(
                    false, ahi.v, false, bhi.v, (short)0, acc[mt], false, false);
                acc[mt] = __builtin_amdgcn_wmma_f32_16x16x32_bf16(
                    false, ahi.v, false, blo.v, (short)0, acc[mt], false, false);
                acc[mt] = __builtin_amdgcn_wmma_f32_16x16x32_bf16(
                    false, alo.v, false, bhi.v, (short)0, acc[mt], false, false);
            }
        }

        async_wait0();      // my copies into buf^1 have landed in LDS
        __syncthreads();    // all waves' copies landed; all done reading buf
        buf ^= 1;
    }

    // C/D layout: VGPR r, lanes 0-15 -> M=r, lanes 16-31 -> M=8+r; N = lane&15
#pragma unroll
    for (int mt = 0; mt < 4; ++mt)
#pragma unroll
        for (int r = 0; r < 8; ++r) {
            int brow = mt * 16 + r + 8 * khalf;
            logits[(size_t)brow * V_ + ncol] = acc[mt][r];
        }
}

// =====================================================================
// Kernel 3: fused softmax / inverse-CDF sample / greedy / top-8.
// One 1024-thread block per row; logits row (512KB) is L2-resident.
// =====================================================================
__global__ __launch_bounds__(1024) void sample_topk_kernel(
    const float* __restrict__ logits, const float* __restrict__ temperature,
    const float* __restrict__ uniform, const int* __restrict__ greedy_mask,
    float* __restrict__ out) {
    const int b = blockIdx.x, t = threadIdx.x;
    const int CH = V_ / 1024;            // 125 contiguous elements per thread
    const int start = t * CH;
    const float* row = logits + (size_t)b * V_;
    const float invT = 1.0f / temperature[b];

    __shared__ unsigned long long red[1024];
    __shared__ float ps[1024];
    __shared__ int   s_sel[TOPK];
    __shared__ int   s_sample;

    // ---- pass 0: block argmax (greedy id, row max) ----
    float best = row[start];
    int   bi   = start;
    for (int i = 1; i < CH; ++i) {
        float v = row[start + i];
        if (v > best) { best = v; bi = start + i; }
    }
    red[t] = argmax_key(best, bi);
    __syncthreads();
    for (int off = 512; off > 0; off >>= 1) {
        if (t < off) { unsigned long long o = red[t + off]; if (o > red[t]) red[t] = o; }
        __syncthreads();
    }
    const unsigned long long k0 = red[0];
    const float lmax = key_val(k0);
    const int   gidx = key_idx(k0);
    __syncthreads();

    // ---- exp sums + inclusive prefix scan (CDF blocks) ----
    float s_local = 0.f;
    for (int i = 0; i < CH; ++i) s_local += __expf((row[start + i] - lmax) * invT);
    ps[t] = s_local;
    __syncthreads();
    for (int off = 1; off < 1024; off <<= 1) {
        float v = (t >= off) ? ps[t - off] : 0.f;
        __syncthreads();
        ps[t] += v;
        __syncthreads();
    }
    const float S    = ps[1023];
    const float logS = __logf(S);

    // ---- inverse-CDF sampling ----
    const float target = uniform[b] * S;
    if (t == 0) s_sample = V_ - 1;
    __syncthreads();
    float pexc = ps[t] - s_local;
    if (pexc < target && ps[t] >= target) {
        float c = pexc;
        for (int i = 0; i < CH; ++i) {
            c += __expf((row[start + i] - lmax) * invT);
            if (c >= target) { s_sample = start + i; break; }
        }
    }
    __syncthreads();

    if (t == 0) {
        s_sel[0] = gidx;
        int nid = (greedy_mask[b] != 0) ? gidx : s_sample;
        out[b]       = (float)nid;                                  // next_token_ids
        out[B_ + b]  = (row[nid] - lmax) * invT - logS;             // chosen_logprobs
        out[2 * B_ + b * TOPK + 0]              = (float)gidx;      // topk_indices[0]
        out[2 * B_ + B_ * TOPK + b * TOPK + 0]  = -logS;            // topk_logprobs[0]
    }
    __syncthreads();

    // ---- top-k passes 1..7: masked block argmax ----
    for (int p = 1; p < TOPK; ++p) {
        float pb = -INFINITY;
        int   pi = start;
        for (int i = 0; i < CH; ++i) {
            int g = start + i;
            bool skip = false;
            for (int j = 0; j < p; ++j) skip |= (s_sel[j] == g);
            float v = row[g];
            if (!skip && v > pb) { pb = v; pi = g; }
        }
        red[t] = argmax_key(pb, pi);
        __syncthreads();
        for (int off = 512; off > 0; off >>= 1) {
            if (t < off) { unsigned long long o = red[t + off]; if (o > red[t]) red[t] = o; }
            __syncthreads();
        }
        if (t == 0) {
            unsigned long long kp = red[0];
            int   idx = key_idx(kp);
            float val = key_val(kp);
            s_sel[p] = idx;
            out[2 * B_ + b * TOPK + p]             = (float)idx;
            out[2 * B_ + B_ * TOPK + b * TOPK + p] = (val - lmax) * invT - logS;
        }
        __syncthreads();
    }
}

extern "C" void kernel_launch(void* const* d_in, const int* in_sizes, int n_in,
                              void* d_out, int out_size, void* d_ws, size_t ws_size,
                              hipStream_t stream) {
    const float* hidden = (const float*)d_in[0];
    const float* temp   = (const float*)d_in[1];
    const float* uni    = (const float*)d_in[2];
    const int*   gmask  = (const int*)d_in[3];
    const float* nw     = (const float*)d_in[4];
    const float* w      = (const float*)d_in[5];
    float* out = (float*)d_out;

    // workspace: logits fp32 [64][128000], then packed bf16 hi/lo activations
    float*    logits = (float*)d_ws;
    unsigned* a_hi   = (unsigned*)((char*)d_ws + (size_t)B_ * V_ * sizeof(float));
    unsigned* a_lo   = a_hi + (size_t)B_ * (H_ / 2);

    rmsnorm_split_kernel<<<B_, 256, 0, stream>>>(hidden, nw, a_hi, a_lo);
    lmhead_gemm_wmma<<<V_ / 128, 256, 0, stream>>>(a_hi, a_lo, w, logits);
    sample_topk_kernel<<<B_, 1024, 0, stream>>>(logits, temp, uni, gmask, out);
}
